// CausalSelfAttention_17867063951717
// MI455X (gfx1250) — matrix operand order repaired and checked by reading, not compiled, not run on hardware
//
#include <hip/hip_runtime.h>

// ---------------------------------------------------------------------------
// Causal self-attention for MI455X (gfx1250, wave32, WMMA bf16 16x16x32).
// Pipeline: f32->bf16 convert, QKV GEMM (LDS double-buffered via async-to-LDS,
// WMMA), flash attention (WMMA + prefetch), output projection GEMM + bias.
// ---------------------------------------------------------------------------

typedef __bf16 bf16_t;
typedef __attribute__((ext_vector_type(16))) __bf16 v16bf;
typedef __attribute__((ext_vector_type(8)))  __bf16 v8bf;
typedef __attribute__((ext_vector_type(4)))  __bf16 v4bf;
typedef __attribute__((ext_vector_type(8)))  float  v8f;

#define N_EMBD 1024
#define SEQ    2048
#define NH     16
#define HD     64
#define BATCH  4

__device__ __forceinline__ v8f zero_v8f() {
  v8f z;
#pragma unroll
  for (int i = 0; i < 8; ++i) z[i] = 0.0f;
  return z;
}

__device__ __forceinline__ v8f wmma_bf16(v16bf a, v16bf b, v8f c) {
  // D = A(16x32 bf16) x B(32x16 bf16) + C(16x16 f32)
  return __builtin_amdgcn_wmma_f32_16x16x32_bf16(
      /*neg_a=*/false, a, /*neg_b=*/false, b,
      /*c_mod=*/(short)0, c, /*reuse_a=*/false, /*reuse_b=*/false);
}

// A-fragment (16x32 bf16): lane m = lane&15.
// lanes 0-15 hold K = [0..15]; lanes 16-31 hold K = [16..31].
__device__ __forceinline__ v16bf load_a_row(const bf16_t* row) {
  const int half = (threadIdx.x >> 4) & 1;
  const bf16_t* p = row + half * 16;
  v8bf lo = *(const v8bf*)(p);
  v8bf hi = *(const v8bf*)(p + 8);
  v16bf a;
#pragma unroll
  for (int i = 0; i < 8; ++i) { a[i] = lo[i]; a[i + 8] = hi[i]; }
  return a;
}

// B-fragment (32x16 bf16): lane n = lane&15; lanes 0-15 hold K=[0..15],
// lanes 16-31 hold K=[16..31]. Column must be K-contiguous (NT layout).
__device__ __forceinline__ v16bf load_b_col(const bf16_t* col) {
  const int half = (threadIdx.x >> 4) & 1;
  return *(const v16bf*)(col + half * 16);
}

// --- CDNA5 async global->LDS copy (ASYNCcnt), 16B per lane -----------------
__device__ __forceinline__ void async_ld16(const bf16_t* g, bf16_t* l) {
  // Generic shared pointer low 32 bits == LDS byte offset on AMDGPU.
  const unsigned lds_off = (unsigned)(size_t)l;
  asm volatile("global_load_async_to_lds_b128 %0, %1, off"
               :: "v"(lds_off), "v"(g)
               : "memory");
}

__device__ __forceinline__ void wait_async0() {
  asm volatile("s_wait_asynccnt 0x0" ::: "memory");
}

// ---------------------------------------------------------------------------
// f32 -> bf16 conversion (vectorized x4)
// ---------------------------------------------------------------------------
__global__ void cvt_f32_bf16(const float* __restrict__ in,
                             bf16_t* __restrict__ out, int n4) {
  const int i = blockIdx.x * blockDim.x + threadIdx.x;
  if (i < n4) {
    const float4 v = ((const float4*)in)[i];
    v4bf o;
    o[0] = (bf16_t)v.x; o[1] = (bf16_t)v.y;
    o[2] = (bf16_t)v.z; o[3] = (bf16_t)v.w;
    ((v4bf*)out)[i] = o;
  }
}

// ---------------------------------------------------------------------------
// LDS-staged NT GEMM machinery: block tile 64(M) x 128(N), K-step 32,
// 8 waves arranged 4(M) x 2(N); each wave computes 16x64 (4 C-frags).
// A slice 64x32 (4KB) + B slice 128x32 (8KB) double-buffered (24KB LDS).
// ---------------------------------------------------------------------------
__device__ __forceinline__ void stage_tiles(const bf16_t* __restrict__ Ag,
                                            const bf16_t* __restrict__ Bg,
                                            bf16_t* Al, bf16_t* Bl,
                                            int k0, int tid) {
  {  // A: 64 rows x 32 K -> 256 16B chunks, one per thread
    const int row = tid >> 2, ch = tid & 3;
    async_ld16(Ag + (size_t)row * N_EMBD + k0 + ch * 8, Al + row * 32 + ch * 8);
  }
#pragma unroll
  for (int i = 0; i < 2; ++i) {  // B: 128 rows x 32 K -> 512 chunks
    const int c = tid + i * 256;
    const int row = c >> 2, ch = c & 3;
    async_ld16(Bg + (size_t)row * N_EMBD + k0 + ch * 8, Bl + row * 32 + ch * 8);
  }
}

__device__ __forceinline__ void compute_step(const bf16_t* Al, const bf16_t* Bl,
                                             int wm, int wn, int lane,
                                             v8f& c0, v8f& c1, v8f& c2, v8f& c3) {
  const v16bf a = load_a_row(Al + (size_t)(wm * 16 + (lane & 15)) * 32);
  const bf16_t* bb = Bl + (size_t)(wn * 64 + (lane & 15)) * 32;
  c0 = wmma_bf16(a, load_b_col(bb + 0 * 16 * 32), c0);
  c1 = wmma_bf16(a, load_b_col(bb + 1 * 16 * 32), c1);
  c2 = wmma_bf16(a, load_b_col(bb + 2 * 16 * 32), c2);
  c3 = wmma_bf16(a, load_b_col(bb + 3 * 16 * 32), c3);
}

// ---------------------------------------------------------------------------
// QKV projection: qkv[m, col] = sum_c x[m,c] * Wqkv[col,c] + b[col]
// Epilogue scatters Q,K row-major [B,H,S,D] and V transposed [B,H,D,S].
// ---------------------------------------------------------------------------
__global__ __launch_bounds__(256) void qkv_gemm_kernel(
    const bf16_t* __restrict__ xb, const bf16_t* __restrict__ wb,
    const float* __restrict__ bias,
    bf16_t* __restrict__ qb, bf16_t* __restrict__ kb,
    bf16_t* __restrict__ vtb) {
  __shared__ __align__(64) bf16_t Asm[2][64 * 32];
  __shared__ __align__(64) bf16_t Bsm[2][128 * 32];

  const int tid = threadIdx.x;
  const int lane = tid & 31;
  const int wave = tid >> 5;
  const int wm = wave & 3;        // 0..3 -> M sub-tile
  const int wn = wave >> 2;       // 0..1 -> N sub-tile
  const int mblock = blockIdx.y * 64;
  const int nblock = blockIdx.x * 128;

  const bf16_t* Ag = xb + (size_t)mblock * N_EMBD;
  const bf16_t* Bg = wb + (size_t)nblock * N_EMBD;

  v8f c0 = zero_v8f(), c1 = zero_v8f(), c2 = zero_v8f(), c3 = zero_v8f();

  stage_tiles(Ag, Bg, Asm[0], Bsm[0], 0, tid);
  wait_async0();
  __syncthreads();

  for (int k0 = 0; k0 < N_EMBD; k0 += 64) {
    stage_tiles(Ag, Bg, Asm[1], Bsm[1], k0 + 32, tid);
    compute_step(Asm[0], Bsm[0], wm, wn, lane, c0, c1, c2, c3);
    wait_async0();
    __syncthreads();
    if (k0 + 64 < N_EMBD)
      stage_tiles(Ag, Bg, Asm[0], Bsm[0], k0 + 64, tid);
    compute_step(Asm[1], Bsm[1], wm, wn, lane, c0, c1, c2, c3);
    wait_async0();
    __syncthreads();
  }

  // ---- epilogue: bias + scatter into Q / K / V^T ----
  const int rowoff = (lane >> 4) * 8;
  const int ncol = lane & 15;
  v8f* accs[4] = {&c0, &c1, &c2, &c3};
#pragma unroll
  for (int j = 0; j < 4; ++j) {
    const int col = nblock + wn * 64 + j * 16 + ncol;
    const int sel = col >> 10;          // 0=Q 1=K 2=V
    const int rem = col & (N_EMBD - 1);
    const int h = rem >> 6;
    const int d = rem & (HD - 1);
    const float bv = bias[col];
#pragma unroll
    for (int r = 0; r < 8; ++r) {
      const int m = mblock + wm * 16 + rowoff + r;
      const int b_ = m >> 11;           // m / SEQ
      const int s = m & (SEQ - 1);
      const int bh = b_ * NH + h;
      const bf16_t v = (bf16_t)((*accs[j])[r] + bv);
      if (sel == 0)      qb[((size_t)bh * SEQ + s) * HD + d] = v;
      else if (sel == 1) kb[((size_t)bh * SEQ + s) * HD + d] = v;
      else               vtb[((size_t)bh * HD + d) * SEQ + s] = v;   // V^T
    }
  }
}

// ---------------------------------------------------------------------------
// Flash attention: grid (B*H, S/128), block = 8 waves; wave owns 16 q rows.
// Per 32-key tile: 4 score WMMAs (QxK^T), online softmax (shuffle reduce),
// P staged through LDS (C-layout -> A-layout), 4 PV WMMAs against V^T.
// ---------------------------------------------------------------------------
__global__ __launch_bounds__(256) void attn_kernel(
    const bf16_t* __restrict__ qb, const bf16_t* __restrict__ kb,
    const bf16_t* __restrict__ vtb, bf16_t* __restrict__ yb) {
  __shared__ __align__(64) bf16_t pbuf[8][16 * 32];

  const int lane = threadIdx.x & 31;
  const int wave = threadIdx.x >> 5;
  const int bh = blockIdx.x;
  const int h = bh & (NH - 1);
  const int bidx = bh >> 4;
  const int qbase = blockIdx.y * 128 + wave * 16;

  const bf16_t* Q    = qb  + (size_t)bh * SEQ * HD;
  const bf16_t* Kmat = kb  + (size_t)bh * SEQ * HD;
  const bf16_t* Vt   = vtb + (size_t)bh * HD * SEQ;

  const int mq = qbase + (lane & 15);
  const v16bf qf0 = load_a_row(Q + (size_t)mq * HD);
  const v16bf qf1 = load_a_row(Q + (size_t)mq * HD + 32);

  v8f o0 = zero_v8f(), o1 = zero_v8f(), o2 = zero_v8f(), o3 = zero_v8f();
  float mrun[8], srun[8];
#pragma unroll
  for (int r = 0; r < 8; ++r) { mrun[r] = -1e30f; srun[r] = 0.0f; }

  const int rowoff = (lane >> 4) * 8;   // C-frag rows this lane holds
  const int ncol = lane & 15;           // C-frag column this lane holds
  bf16_t* pb = pbuf[wave];

  const int ntiles = (qbase + 47) >> 5;           // causal key-tile count
  for (int kt = 0; kt < ntiles; ++kt) {
    const int kbase = kt * 32;

    // prefetch next key tile's K and V^T streams while computing this one
    if (kt + 1 < ntiles) {
      __builtin_prefetch(Kmat + (size_t)(kbase + 32 + ncol) * HD, 0, 0);
      __builtin_prefetch(Kmat + (size_t)(kbase + 48 + ncol) * HD, 0, 0);
      __builtin_prefetch(Vt + (size_t)ncol * SEQ + kbase + 32, 0, 0);
      __builtin_prefetch(Vt + (size_t)(32 + ncol) * SEQ + kbase + 32, 0, 0);
    }

    // ---- scores = Q (16x64) x K^T (64x32), two 16x16 C-frags ----
    v8f s0 = zero_v8f(), s1 = zero_v8f();
    {
      const bf16_t* kc0 = Kmat + (size_t)(kbase + ncol) * HD;
      const bf16_t* kc1 = Kmat + (size_t)(kbase + 16 + ncol) * HD;
      s0 = wmma_bf16(qf0, load_b_col(kc0),      s0);
      s0 = wmma_bf16(qf1, load_b_col(kc0 + 32), s0);
      s1 = wmma_bf16(qf0, load_b_col(kc1),      s1);
      s1 = wmma_bf16(qf1, load_b_col(kc1 + 32), s1);
    }

    // ---- online softmax (scale, causal mask, row max/sum via shuffles) ----
    const float sc = 0.125f;  // 1/sqrt(64)
#pragma unroll
    for (int r = 0; r < 8; ++r) {
      const int q = qbase + rowoff + r;
      float v0 = s0[r] * sc;
      float v1 = s1[r] * sc;
      if (kbase + ncol > q)      v0 = -1e30f;
      if (kbase + 16 + ncol > q) v1 = -1e30f;
      float mx = fmaxf(v0, v1);
#pragma unroll
      for (int off = 1; off < 16; off <<= 1)
        mx = fmaxf(mx, __shfl_xor(mx, off, 32));
      const float mnew  = fmaxf(mrun[r], mx);
      const float alpha = __expf(mrun[r] - mnew);
      const float p0 = __expf(v0 - mnew);
      const float p1 = __expf(v1 - mnew);
      float ps = p0 + p1;
#pragma unroll
      for (int off = 1; off < 16; off <<= 1)
        ps += __shfl_xor(ps, off, 32);
      srun[r] = srun[r] * alpha + ps;
      mrun[r] = mnew;
      o0[r] *= alpha; o1[r] *= alpha; o2[r] *= alpha; o3[r] *= alpha;
      // stage P (bf16) to wave-private LDS in [row][key] order
      pb[(rowoff + r) * 32 + ncol]      = (bf16_t)p0;
      pb[(rowoff + r) * 32 + 16 + ncol] = (bf16_t)p1;
    }

    asm volatile("" ::: "memory");
    const v16bf pf = load_a_row(pb + (size_t)(lane & 15) * 32); // A-layout P
    asm volatile("" ::: "memory");

    // ---- O += P (16x32) x V (32x64), V^T columns are key-contiguous ----
    {
      const bf16_t* v0c = Vt + (size_t)(ncol)      * SEQ + kbase;
      const bf16_t* v1c = Vt + (size_t)(16 + ncol) * SEQ + kbase;
      const bf16_t* v2c = Vt + (size_t)(32 + ncol) * SEQ + kbase;
      const bf16_t* v3c = Vt + (size_t)(48 + ncol) * SEQ + kbase;
      o0 = wmma_bf16(pf, load_b_col(v0c), o0);
      o1 = wmma_bf16(pf, load_b_col(v1c), o1);
      o2 = wmma_bf16(pf, load_b_col(v2c), o2);
      o3 = wmma_bf16(pf, load_b_col(v3c), o3);
    }
  }

  // ---- normalize and write y[b, s, h*64 + d] as bf16 ----
#pragma unroll
  for (int r = 0; r < 8; ++r) {
    const float inv = 1.0f / srun[r];
    const int s = qbase + rowoff + r;
    const size_t rowbase = ((size_t)bidx * SEQ + s) * N_EMBD + (size_t)h * HD;
    yb[rowbase + ncol]      = (bf16_t)(o0[r] * inv);
    yb[rowbase + 16 + ncol] = (bf16_t)(o1[r] * inv);
    yb[rowbase + 32 + ncol] = (bf16_t)(o2[r] * inv);
    yb[rowbase + 48 + ncol] = (bf16_t)(o3[r] * inv);
  }
}

// ---------------------------------------------------------------------------
// Output projection: out[m, col] = sum_c y[m,c] * Wout[col,c] + b[col]  (f32)
// ---------------------------------------------------------------------------
__global__ __launch_bounds__(256) void out_gemm_kernel(
    const bf16_t* __restrict__ yb, const bf16_t* __restrict__ wb,
    const float* __restrict__ bias, float* __restrict__ out) {
  __shared__ __align__(64) bf16_t Asm[2][64 * 32];
  __shared__ __align__(64) bf16_t Bsm[2][128 * 32];

  const int tid = threadIdx.x;
  const int lane = tid & 31;
  const int wave = tid >> 5;
  const int wm = wave & 3;
  const int wn = wave >> 2;
  const int mblock = blockIdx.y * 64;
  const int nblock = blockIdx.x * 128;

  const bf16_t* Ag = yb + (size_t)mblock * N_EMBD;
  const bf16_t* Bg = wb + (size_t)nblock * N_EMBD;

  v8f c0 = zero_v8f(), c1 = zero_v8f(), c2 = zero_v8f(), c3 = zero_v8f();

  stage_tiles(Ag, Bg, Asm[0], Bsm[0], 0, tid);
  wait_async0();
  __syncthreads();

  for (int k0 = 0; k0 < N_EMBD; k0 += 64) {
    stage_tiles(Ag, Bg, Asm[1], Bsm[1], k0 + 32, tid);
    compute_step(Asm[0], Bsm[0], wm, wn, lane, c0, c1, c2, c3);
    wait_async0();
    __syncthreads();
    if (k0 + 64 < N_EMBD)
      stage_tiles(Ag, Bg, Asm[0], Bsm[0], k0 + 64, tid);
    compute_step(Asm[1], Bsm[1], wm, wn, lane, c0, c1, c2, c3);
    wait_async0();
    __syncthreads();
  }

  const int rowoff = (lane >> 4) * 8;
  const int ncol = lane & 15;
  v8f* accs[4] = {&c0, &c1, &c2, &c3};
#pragma unroll
  for (int j = 0; j < 4; ++j) {
    const int col = nblock + wn * 64 + j * 16 + ncol;
    const float bv = bias[col];
#pragma unroll
    for (int r = 0; r < 8; ++r) {
      const int m = mblock + wm * 16 + rowoff + r;
      out[(size_t)m * N_EMBD + col] = (*accs[j])[r] + bv;
    }
  }
}

// ---------------------------------------------------------------------------
// Launch
// ---------------------------------------------------------------------------
extern "C" void kernel_launch(void* const* d_in, const int* in_sizes, int n_in,
                              void* d_out, int out_size, void* d_ws, size_t ws_size,
                              hipStream_t stream) {
  (void)in_sizes; (void)n_in; (void)out_size; (void)ws_size;

  const float* x     = (const float*)d_in[0];
  const float* W_qkv = (const float*)d_in[1];
  const float* b_qkv = (const float*)d_in[2];
  const float* W_out = (const float*)d_in[3];
  const float* b_out = (const float*)d_in[4];
  float* out = (float*)d_out;

  const size_t n_x    = (size_t)BATCH * SEQ * N_EMBD;  // 8,388,608
  const size_t n_wqkv = (size_t)3 * N_EMBD * N_EMBD;   // 3,145,728
  const size_t n_wout = (size_t)N_EMBD * N_EMBD;       // 1,048,576

  bf16_t* xb    = (bf16_t*)d_ws;
  bf16_t* wqkvb = xb + n_x;
  bf16_t* woutb = wqkvb + n_wqkv;
  bf16_t* qb    = woutb + n_wout;
  bf16_t* kb    = qb + n_x;
  bf16_t* vtb   = kb + n_x;
  bf16_t* yb    = vtb + n_x;

  const int thr = 256;
  cvt_f32_bf16<<<(int)((n_x / 4 + thr - 1) / thr), thr, 0, stream>>>(
      x, xb, (int)(n_x / 4));
  cvt_f32_bf16<<<(int)((n_wqkv / 4 + thr - 1) / thr), thr, 0, stream>>>(
      W_qkv, wqkvb, (int)(n_wqkv / 4));
  cvt_f32_bf16<<<(int)((n_wout / 4 + thr - 1) / thr), thr, 0, stream>>>(
      W_out, woutb, (int)(n_wout / 4));

  // Block tile 64x128: M=8192 -> 128 tiles; N=3072 -> 24 tiles.
  qkv_gemm_kernel<<<dim3(24, 128), 256, 0, stream>>>(xb, wqkvb, b_qkv,
                                                     qb, kb, vtb);
  // (B*H = 64, S/128 = 16)
  attn_kernel<<<dim3(BATCH * NH, SEQ / 128), 256, 0, stream>>>(qb, kb, vtb, yb);
  // N=1024 -> 8 tiles.
  out_gemm_kernel<<<dim3(8, 128), 256, 0, stream>>>(yb, woutb, b_out, out);
}
